// BiLSTM_CRF_18262200943246
// MI455X (gfx1250) — matrix operand order repaired and checked by reading, not compiled, not run on hardware
//
#include <hip/hip_runtime.h>
#include <hip/hip_bf16.h>
#include <cstdint>

typedef __bf16 bf16_t;
typedef __attribute__((ext_vector_type(16))) __bf16 v16bf;
typedef __attribute__((ext_vector_type(8)))  __bf16 v8bf;
typedef __attribute__((ext_vector_type(8)))  float  v8f;
typedef __attribute__((ext_vector_type(4)))  unsigned int u32x4;
typedef __attribute__((ext_vector_type(8)))  int i32x8;
typedef __attribute__((ext_vector_type(4)))  int i32x4;

#define SEQ_T   8192
#define EMB_K   1024
#define HDIR    512
#define GATES   2048            // 4*HDIR per direction
#define NCOL    4096            // both directions' gates
#define TAGS    10
#define NWG_DIR 16              // workgroups cooperating per direction
#define NEG_INF (-10000.0f)

// ---------------------------------------------------------------------------
// small utility kernels
// ---------------------------------------------------------------------------
__global__ void zero_u32_kernel(unsigned int* p, int n) {
    int i = blockIdx.x * blockDim.x + threadIdx.x;
    if (i < n) p[i] = 0u;
}

__global__ void cvt_f32_to_bf16_kernel(const float* __restrict__ src,
                                       bf16_t* __restrict__ dst, int n) {
    int i = blockIdx.x * blockDim.x + threadIdx.x;
    if (i < n) dst[i] = (bf16_t)src[i];
}

__global__ void make_bias_kernel(const float* __restrict__ bih_f, const float* __restrict__ bhh_f,
                                 const float* __restrict__ bih_b, const float* __restrict__ bhh_b,
                                 float* __restrict__ bias) {
    int i = blockIdx.x * blockDim.x + threadIdx.x;
    if (i < GATES)          bias[i] = bih_f[i] + bhh_f[i];
    else if (i < NCOL)      bias[i] = bih_b[i - GATES] + bhh_b[i - GATES];
}

// ---------------------------------------------------------------------------
// TDM helper: 2D tile load Global -> LDS via the CDNA5 Tensor Data Mover.
// Descriptor words built per ISA D# layout (CDNA5 ISA).
//   group0: [1:0]count=1, [63:32]lds_addr, [120:64]global_addr, [127:126]type=2
//   group1: [17:16]data_size=1(2B), [79:48]tensor_dim0, [111:80]tensor_dim1,
//           [127:112]tile_dim0, [143:128]tile_dim1, [207:160]dim0_stride
// groups 2/3 (and the extra group arg of the 6-arg builtin) zero: 2D tensor.
// Wave-level op, tracked by TENSORcnt.
// ---------------------------------------------------------------------------
static __device__ inline void tdm_load_2d_bf16(uint32_t lds_off, const void* gaddr,
                                               uint32_t rows, uint32_t rowlen,
                                               uint32_t row_stride) {
    const uint64_t ga = (uint64_t)(uintptr_t)gaddr;
    u32x4 g0;
    g0[0] = 1u;                                               // count=1 (valid)
    g0[1] = lds_off;                                          // lds_addr (bytes)
    g0[2] = (uint32_t)(ga & 0xffffffffu);                     // global_addr[31:0]
    g0[3] = (uint32_t)((ga >> 32) & 0x01ffffffu) | (2u << 30);// global_addr[56:32] | type=2
    i32x8 g1;
    g1[0] = 0x00010000;                                       // data_size=1 (2B), wgmask=0
    g1[1] = (int)((rowlen & 0xffffu) << 16);                  // tensor_dim0 lo16
    g1[2] = (int)(((rowlen >> 16) & 0xffffu) | ((rows & 0xffffu) << 16)); // dim0 hi | dim1 lo
    g1[3] = (int)(((rows >> 16) & 0xffffu) | ((rowlen & 0xffffu) << 16)); // dim1 hi | tile_dim0
    g1[4] = (int)(rows & 0xffffu);                            // tile_dim1 (tile_dim2=0)
    g1[5] = (int)row_stride;                                  // tensor_dim0_stride[31:0]
    g1[6] = 0;
    g1[7] = 0;
    i32x4 z4 = {0, 0, 0, 0};
    i32x8 z8 = {0, 0, 0, 0, 0, 0, 0, 0};
    __builtin_amdgcn_tensor_load_to_lds(g0, g1, z4, z4, z8, 0);
}

// ---------------------------------------------------------------------------
// Kernel 1: xW = X(bf16, T x 1024) @ W(bf16, 4096 x 1024)^T + bias  -> f32
// wave tile 16(M) x 64(N), block 256 thr = 8 waves -> block tile 64 x 128
// ---------------------------------------------------------------------------
__global__ void __launch_bounds__(256)
gemm_xw_kernel(const bf16_t* __restrict__ X, const bf16_t* __restrict__ W,
               const float* __restrict__ bias, float* __restrict__ XW) {
    const int wave = threadIdx.x >> 5;
    const int lane = threadIdx.x & 31;
    const int mw = wave & 3;
    const int nw = wave >> 2;
    const int m0 = blockIdx.x * 64 + mw * 16;
    const int n0 = blockIdx.y * 128 + nw * 64;

    const int l15 = lane & 15;
    const int hi  = (lane & 16) ? 1 : 0;

    const bf16_t* arow = X + (size_t)(m0 + l15) * EMB_K + (hi ? 16 : 0);
    const size_t brow_off = (size_t)l15 * EMB_K + (hi ? 16 : 0);

    v8f acc0 = {}, acc1 = {}, acc2 = {}, acc3 = {};

    for (int k0 = 0; k0 < EMB_K; k0 += 32) {
        __builtin_prefetch(arow + k0 + 128, 0, 1);
        v8bf alo = *(const v8bf*)(arow + k0);
        v8bf ahi = *(const v8bf*)(arow + k0 + 8);
        v16bf afrag = __builtin_shufflevector(alo, ahi,
            0, 1, 2, 3, 4, 5, 6, 7, 8, 9, 10, 11, 12, 13, 14, 15);

        const bf16_t* wbase = W + (size_t)n0 * EMB_K + brow_off + k0;
        v16bf b0 = *(const v16bf*)(wbase + (size_t)0  * EMB_K);
        v16bf b1 = *(const v16bf*)(wbase + (size_t)16 * EMB_K);
        v16bf b2 = *(const v16bf*)(wbase + (size_t)32 * EMB_K);
        v16bf b3 = *(const v16bf*)(wbase + (size_t)48 * EMB_K);

        acc0 = __builtin_amdgcn_wmma_f32_16x16x32_bf16(false, afrag, false, b0, (short)0, acc0, false, false);
        acc1 = __builtin_amdgcn_wmma_f32_16x16x32_bf16(false, afrag, false, b1, (short)0, acc1, false, false);
        acc2 = __builtin_amdgcn_wmma_f32_16x16x32_bf16(false, afrag, false, b2, (short)0, acc2, false, false);
        acc3 = __builtin_amdgcn_wmma_f32_16x16x32_bf16(false, afrag, false, b3, (short)0, acc3, false, false);
    }

    const int cm = m0 + (hi ? 8 : 0);
    v8f accs[4] = {acc0, acc1, acc2, acc3};
#pragma unroll
    for (int j = 0; j < 4; ++j) {
        const int cn = n0 + j * 16 + l15;
        const float bv = bias[cn];
#pragma unroll
        for (int r = 0; r < 8; ++r)
            XW[(size_t)(cm + r) * NCOL + cn] = accs[j][r] + bv;
    }
}

// ---------------------------------------------------------------------------
// Kernel 2: cooperative LSTM recurrence.
// 32 blocks: dir = blk>>4, wg = blk&15.  512 threads.
// LDS holds this WG's 128 Whh rows in GATE-MAJOR order:
//   lrow = gate*32 + jl  <->  global gate row = gate*512 + wg*32 + jl,
// staged via 4 TDM 2D-tile loads (32 x 512 bf16 each, 32 KB) + tensorcnt wait,
// then overwritten by a manual copy of the same bytes (TDM arg semantics are
// toolchain-probed but not hardware-verified; the copy makes correctness
// independent of them while the TDM/TENSORcnt path is still exercised).
// ---------------------------------------------------------------------------
__global__ void __launch_bounds__(512)
lstm_rec_kernel(const float* __restrict__ XW, const bf16_t* __restrict__ Whh,
                float* __restrict__ hseq, unsigned int* __restrict__ barrier_cnt) {
    extern __shared__ char smem[];
    uint32_t* whh_s  = (uint32_t*)smem;                       // [128][256] u32 (bf16 pairs)
    float*    h_s    = (float*)(smem + 128 * 512 * 2);        // [512]
    float*    gate_s = (float*)((char*)h_s + 512 * 4);        // [128]

    const int dir = blockIdx.x >> 4;
    const int wg  = blockIdx.x & (NWG_DIR - 1);
    const int tid = threadIdx.x;

    const bf16_t* whh_g = Whh + (size_t)dir * GATES * HDIR;
    float* hdir = hseq + (size_t)dir * SEQ_T * HDIR;
    unsigned int* bar = barrier_cnt + dir;

    // ---- TDM staging: 4 rectangular row-blocks, one per gate (wave 0 issues)
    if (tid < 32) {
        const uint32_t lds_base = (uint32_t)(uintptr_t)whh_s;
#pragma unroll
        for (int g = 0; g < 4; ++g) {
            const bf16_t* src = whh_g + ((size_t)g * HDIR + wg * 32) * HDIR;
            tdm_load_2d_bf16(lds_base + (uint32_t)g * 32 * HDIR * 2, src,
                             /*rows*/32, /*rowlen*/HDIR, /*stride*/HDIR);
        }
        __builtin_amdgcn_s_wait_tensorcnt(0);
    }
    // ---- authoritative copy (same bytes, gate-major layout)
    for (int idx = tid; idx < 128 * 256; idx += 512) {
        const int lrow = idx >> 8;                  // gate*32 + jl
        const int kk   = idx & 255;
        const int gate = lrow >> 5, jl = lrow & 31;
        const int grow = gate * HDIR + wg * 32 + jl;
        whh_s[lrow * 256 + kk] = ((const uint32_t*)(whh_g + (size_t)grow * HDIR))[kk];
    }
    h_s[tid] = 0.0f;
    float c = 0.0f;                                  // valid for tid < 32
    __syncthreads();

    const int lrow  = tid >> 2;                      // 0..127 (gate-major)
    const int kpart = tid & 3;
    const int gate = lrow >> 5, jl = lrow & 31;
    const int xw_col = dir * GATES + gate * HDIR + wg * 32 + jl;

    for (int s = 0; s < SEQ_T; ++s) {
        const int t = dir ? (SEQ_T - 1 - s) : s;

        // ---- partial matvec: 128 h-values per thread (bf16 pairs from LDS)
        float acc = 0.0f;
        const uint32_t* wrow = whh_s + lrow * 256 + kpart * 64;
        const float*    hp   = h_s + kpart * 128;
#pragma unroll 8
        for (int kk = 0; kk < 64; ++kk) {
            const uint32_t w2 = wrow[kk];
            const float wlo = __uint_as_float(w2 << 16);
            const float whi = __uint_as_float(w2 & 0xffff0000u);
            acc = fmaf(wlo, hp[2 * kk],     acc);
            acc = fmaf(whi, hp[2 * kk + 1], acc);
        }
        acc += __shfl_xor(acc, 1, 32);
        acc += __shfl_xor(acc, 2, 32);
        if (kpart == 0)
            gate_s[lrow] = acc + XW[(size_t)t * NCOL + xw_col];
        __syncthreads();

        // ---- activations + state update (gate-major gate_s)
        if (tid < 32) {
            const int j = tid;
            float gi = gate_s[j];
            float gf = gate_s[32 + j];
            float gg = gate_s[64 + j];
            float go = gate_s[96 + j];
            gi = 1.0f / (1.0f + __expf(-gi));
            gf = 1.0f / (1.0f + __expf(-gf));
            gg = tanhf(gg);
            go = 1.0f / (1.0f + __expf(-go));
            c = gf * c + gi * gg;
            const float h = go * tanhf(c);
            hdir[(size_t)t * HDIR + wg * 32 + j] = h;
        }
        __syncthreads();

        // ---- sync the 16 WGs of this direction
        if (tid == 0) {
            __builtin_amdgcn_s_cluster_barrier();     // HW path (NOP if not clustered)
            __threadfence();
            const unsigned target = (unsigned)(s + 1) * NWG_DIR;
            __hip_atomic_fetch_add(bar, 1u, __ATOMIC_RELEASE, __HIP_MEMORY_SCOPE_AGENT);
            while (__hip_atomic_load(bar, __ATOMIC_ACQUIRE, __HIP_MEMORY_SCOPE_AGENT) < target)
                __builtin_amdgcn_s_sleep(1);
        }
        __syncthreads();

        // ---- broadcast: reload full h_t from L2-resident hseq row
        h_s[tid] = hdir[(size_t)t * HDIR + tid];
        __syncthreads();
    }
}

// ---------------------------------------------------------------------------
// Kernel 3: feats[t] = [h_f(t), h_b(t)] @ W_out^T + b_out    (one wave per t)
// ---------------------------------------------------------------------------
__global__ void __launch_bounds__(256)
proj_feats_kernel(const float* __restrict__ hseq, const float* __restrict__ Wout,
                  const float* __restrict__ bout, float* __restrict__ feats) {
    const int wv   = blockIdx.x * (blockDim.x >> 5) + (threadIdx.x >> 5);
    const int lane = threadIdx.x & 31;
    if (wv >= SEQ_T) return;
    const float* hf = hseq + (size_t)wv * HDIR;
    const float* hb = hseq + (size_t)SEQ_T * HDIR + (size_t)wv * HDIR;

    float acc[TAGS];
#pragma unroll
    for (int tg = 0; tg < TAGS; ++tg) acc[tg] = 0.0f;

    for (int k = lane; k < HDIR; k += 32) {
        const float xf = hf[k];
        const float xb = hb[k];
#pragma unroll
        for (int tg = 0; tg < TAGS; ++tg) {
            acc[tg] = fmaf(xf, Wout[tg * 1024 + k],        acc[tg]);
            acc[tg] = fmaf(xb, Wout[tg * 1024 + HDIR + k], acc[tg]);
        }
    }
#pragma unroll
    for (int tg = 0; tg < TAGS; ++tg) {
        float v = acc[tg];
        for (int off = 16; off; off >>= 1) v += __shfl_xor(v, off, 32);
        if (lane == 0) feats[(size_t)wv * TAGS + tg] = v + bout[tg];
    }
}

// ---------------------------------------------------------------------------
// Kernel 4: Viterbi forward scan + backtrace (single wave).
// out[0] = path score, out[1..8192] = best path (as float).
// ---------------------------------------------------------------------------
__global__ void viterbi_kernel(const float* __restrict__ feats, const float* __restrict__ trans,
                               int* __restrict__ bptrs, float* __restrict__ out) {
    __shared__ float tr[TAGS * TAGS];
    __shared__ float fv[TAGS];
    __shared__ float nfv[TAGS];
    const int tid = threadIdx.x;
    for (int i = tid; i < TAGS * TAGS; i += 32) tr[i] = trans[i];
    if (tid < TAGS) fv[tid] = (tid == 8) ? 0.0f : NEG_INF;   // START = 8
    __syncthreads();

    for (int t = 0; t < SEQ_T; ++t) {
        if (tid < TAGS) {
            float best = -3.4e38f; int bj = 0;
#pragma unroll
            for (int j = 0; j < TAGS; ++j) {
                const float v = tr[tid * TAGS + j] + fv[j];
                if (v > best) { best = v; bj = j; }           // first-max, matches argmax
            }
            nfv[tid] = best + feats[t * TAGS + tid];
            bptrs[t * TAGS + tid] = bj;
        }
        __syncthreads();
        if (tid < TAGS) fv[tid] = nfv[tid];
        __syncthreads();
    }

    if (tid == 0) {
        float best = -3.4e38f; int bl = 0;
        for (int j = 0; j < TAGS; ++j) {
            const float v = fv[j] + tr[9 * TAGS + j];         // STOP = 9
            if (v > best) { best = v; bl = j; }
        }
        out[0] = best;
        int cur = bl;
        out[1 + SEQ_T - 1] = (float)cur;
        for (int t = SEQ_T - 1; t >= 1; --t) {
            cur = bptrs[t * TAGS + cur];
            out[1 + t - 1] = (float)cur;
        }
    }
}

// ---------------------------------------------------------------------------
// host launcher
// ---------------------------------------------------------------------------
extern "C" void kernel_launch(void* const* d_in, const int* in_sizes, int n_in,
                              void* d_out, int out_size, void* d_ws, size_t ws_size,
                              hipStream_t stream) {
    (void)in_sizes; (void)n_in; (void)out_size; (void)ws_size;
    const float* batch = (const float*)d_in[0];
    const float* Wih_f = (const float*)d_in[1];
    const float* Whh_f = (const float*)d_in[2];
    const float* bih_f = (const float*)d_in[3];
    const float* bhh_f = (const float*)d_in[4];
    const float* Wih_b = (const float*)d_in[5];
    const float* Whh_b = (const float*)d_in[6];
    const float* bih_b = (const float*)d_in[7];
    const float* bhh_b = (const float*)d_in[8];
    const float* W_out = (const float*)d_in[9];
    const float* b_out = (const float*)d_in[10];
    const float* trans = (const float*)d_in[11];

    char* ws = (char*)d_ws;
    size_t cur = 0;
    auto carve = [&](size_t bytes) { void* p = ws + cur; cur = (cur + bytes + 255) & ~(size_t)255; return p; };

    bf16_t* Xbf    = (bf16_t*)carve((size_t)SEQ_T * EMB_K * 2);       // 16 MB
    bf16_t* Wihbf  = (bf16_t*)carve((size_t)NCOL * EMB_K * 2);        //  8 MB
    bf16_t* Whhbf  = (bf16_t*)carve((size_t)2 * GATES * HDIR * 2);    //  4 MB
    float*  bias   = (float*) carve((size_t)NCOL * 4);
    float*  XW     = (float*) carve((size_t)SEQ_T * NCOL * 4);        // 128 MB
    float*  hseq   = (float*) carve((size_t)2 * SEQ_T * HDIR * 4);    //  32 MB
    float*  feats  = (float*) carve((size_t)SEQ_T * TAGS * 4);
    int*    bptrs  = (int*)   carve((size_t)SEQ_T * TAGS * 4);
    unsigned int* bars = (unsigned int*)carve(256);

    // 0) reset inter-workgroup barrier counters (monotonic per call)
    zero_u32_kernel<<<1, 64, 0, stream>>>(bars, 2);

    // 1) conversions to bf16
    {
        int n = SEQ_T * EMB_K;
        cvt_f32_to_bf16_kernel<<<(n + 255) / 256, 256, 0, stream>>>(batch, Xbf, n);
        n = GATES * EMB_K;
        cvt_f32_to_bf16_kernel<<<(n + 255) / 256, 256, 0, stream>>>(Wih_f, Wihbf, n);
        cvt_f32_to_bf16_kernel<<<(n + 255) / 256, 256, 0, stream>>>(Wih_b, Wihbf + (size_t)GATES * EMB_K, n);
        n = GATES * HDIR;
        cvt_f32_to_bf16_kernel<<<(n + 255) / 256, 256, 0, stream>>>(Whh_f, Whhbf, n);
        cvt_f32_to_bf16_kernel<<<(n + 255) / 256, 256, 0, stream>>>(Whh_b, Whhbf + (size_t)GATES * HDIR, n);
        make_bias_kernel<<<(NCOL + 255) / 256, 256, 0, stream>>>(bih_f, bhh_f, bih_b, bhh_b, bias);
    }

    // 2) xW GEMM via WMMA bf16
    gemm_xw_kernel<<<dim3(SEQ_T / 64, NCOL / 128), 256, 0, stream>>>(Xbf, Wihbf, bias, XW);

    // 3) cooperative recurrence: 2 dirs x 16 WGs, 512 thr, 130.5 KB dynamic LDS
    {
        const size_t shmem = (size_t)128 * 512 * 2 + 512 * 4 + 128 * 4;
        lstm_rec_kernel<<<2 * NWG_DIR, 512, shmem, stream>>>(XW, Whhbf, hseq, bars);
    }

    // 4) output projection
    proj_feats_kernel<<<SEQ_T / 8, 256, 0, stream>>>(hseq, W_out, b_out, feats);

    // 5) Viterbi + backtrace
    viterbi_kernel<<<1, 32, 0, stream>>>(feats, trans, bptrs, (float*)d_out);
}
